// RandomAssignMoE_26362509263286
// MI455X (gfx1250) — matrix operand order repaired and checked by reading, not compiled, not run on hardware
//
#include <hip/hip_runtime.h>
#include <hip/hip_bf16.h>
#include <math.h>

// Problem constants (from reference): B=2, S=2048, D=1024, F=2048, E=16
static constexpr int kN = 4096;   // B*S tokens
static constexpr int kD = 1024;
static constexpr int kF = 2048;
static constexpr int kE = 16;

typedef __attribute__((ext_vector_type(16))) __bf16 v16bf;
typedef __attribute__((ext_vector_type(8)))  __bf16 v8bf;
typedef __attribute__((ext_vector_type(8)))  float  v8f;

// ---------------------------------------------------------------------------
// Kernel 1: router.  One wave32 per token: logits = x[n] . router_w[:,e],
// first-occurrence argmax, sigmoid gate, scatter token into expert bucket.
// ---------------------------------------------------------------------------
__global__ __launch_bounds__(256)
void moe_router(const float* __restrict__ x, const float* __restrict__ rw,
                float* __restrict__ wgt, int* __restrict__ counts,
                int* __restrict__ perm)
{
    const int n    = blockIdx.x * 8 + (threadIdx.x >> 5);
    const int lane = threadIdx.x & 31;

    float acc[kE];
#pragma unroll
    for (int e = 0; e < kE; ++e) acc[e] = 0.0f;

    const float* xr = x + (size_t)n * kD;
    for (int d = lane; d < kD; d += 32) {
        const float xv = xr[d];
        const float4* r4 = (const float4*)(rw + (size_t)d * kE);
#pragma unroll
        for (int q = 0; q < 4; ++q) {
            const float4 v = r4[q];
            acc[4*q+0] += xv * v.x;
            acc[4*q+1] += xv * v.y;
            acc[4*q+2] += xv * v.z;
            acc[4*q+3] += xv * v.w;
        }
    }
#pragma unroll
    for (int off = 16; off > 0; off >>= 1) {
#pragma unroll
        for (int e = 0; e < kE; ++e)
            acc[e] += __shfl_xor(acc[e], off, 32);
    }
    if (lane == 0) {
        int best = 0; float bv = acc[0];
#pragma unroll
        for (int e = 1; e < kE; ++e)
            if (acc[e] > bv) { bv = acc[e]; best = e; }   // strict > : first max, like jnp.argmax
        wgt[n] = 1.0f / (1.0f + expf(-bv));
        const int slot = atomicAdd(&counts[best], 1);
        perm[best * kN + slot] = n;
    }
}

// ---------------------------------------------------------------------------
// Grouped GEMM tiles: block = 256 threads (8 waves).  M=32 tokens, N=128,
// K-chunks of 32.  Wave w owns cols [16w,16w+16); two 16x16 C tiles per wave
// reuse one B fragment.  A fragments load as 2x ds_load_b128 (ISA layout:
// lane(h,m) needs lA[m][8h..8h+7] and lA[m][16+8h..16+8h+7], contiguous).
// ---------------------------------------------------------------------------
#define TILE_M 32
#define TILE_N 128
#define TILE_K 32
#define LA_S   (TILE_K + 8)     // 40 bf16 -> 80 B rows, 16B aligned
#define LB_S   (TILE_N + 4)

__device__ __forceinline__ float gelu_erf(float v)
{
    return 0.5f * v * (1.0f + erff(v * 0.70710678118654752f));
}

__device__ __forceinline__ v16bf frag_a(const __bf16* rowp, int half)
{
    const v8bf lo = *(const v8bf*)(rowp + 8 * half);
    const v8bf hi = *(const v8bf*)(rowp + 16 + 8 * half);
    return __builtin_shufflevector(lo, hi, 0,1,2,3,4,5,6,7,8,9,10,11,12,13,14,15);
}

__global__ __launch_bounds__(256)
void moe_gemm1(const float* __restrict__ x, const float* __restrict__ w1,
               const float* __restrict__ b1, const int* __restrict__ counts,
               const int* __restrict__ perm, __bf16* __restrict__ hh)
{
    const int e  = blockIdx.x;          // expert
    const int ft = blockIdx.y;          // F tile (F/128)
    const int mt = blockIdx.z;          // token tile (<= N/32)
    const int cnt = counts[e];
    if (mt * TILE_M >= cnt) return;

    __shared__ __attribute__((aligned(16))) __bf16 lA[TILE_M][LA_S];
    __shared__ __attribute__((aligned(16))) __bf16 lB[TILE_K][LB_S];
    __shared__ int ltok[TILE_M];

    const int tid  = threadIdx.x;
    const int wave = tid >> 5;
    const int lane = tid & 31;
    const int half = lane >> 4;
    const int m    = lane & 15;

    if (tid < TILE_M) {
        int idx = mt * TILE_M + tid;
        if (idx >= cnt) idx = cnt - 1;  // clamp: padded rows duplicate, never stored
        ltok[tid] = perm[e * kN + idx];
    }
    __syncthreads();

    // hoisted per-thread staging pointers
    const int arow = tid >> 3, acol = (tid & 7) * 4;       // A: 32x32, 4 fp32/thread
    const int brow = tid >> 5, bcol = (tid & 31) * 4;      // B: 32x128, 16 fp32/thread
    const float* asrc = x + (size_t)ltok[arow] * kD + acol;
    const float* bsrc = w1 + (size_t)e * kD * kF + (size_t)brow * kF + ft * TILE_N + bcol;

    v8f acc0 = {0.f,0.f,0.f,0.f,0.f,0.f,0.f,0.f};
    v8f acc1 = acc0;

    for (int k0 = 0; k0 < kD; k0 += TILE_K) {
        // --- stage A: 32x32 fp32 -> bf16 (gathered token rows) ---
        {
            const float4 v = *(const float4*)(asrc + k0);
            lA[arow][acol+0] = (__bf16)v.x;  lA[arow][acol+1] = (__bf16)v.y;
            lA[arow][acol+2] = (__bf16)v.z;  lA[arow][acol+3] = (__bf16)v.w;
        }
        // --- stage B: 32x128 fp32 -> bf16, coalesced along F ---
#pragma unroll
        for (int rr = 0; rr < TILE_K; rr += 8) {
            const float4 v = *(const float4*)(bsrc + (size_t)(k0 + rr) * kF);
            lB[brow + rr][bcol+0] = (__bf16)v.x;  lB[brow + rr][bcol+1] = (__bf16)v.y;
            lB[brow + rr][bcol+2] = (__bf16)v.z;  lB[brow + rr][bcol+3] = (__bf16)v.w;
        }
        // prefetch next K-chunk while this one computes
        if (k0 + TILE_K < kD) {
            __builtin_prefetch(asrc + k0 + TILE_K, 0, 3);
            __builtin_prefetch(bsrc + (size_t)(k0 + TILE_K) * kF, 0, 3);
        }
        __syncthreads();

        // --- fragments + WMMA ---
        const v16bf afr0 = frag_a(&lA[m][0],      half);
        const v16bf afr1 = frag_a(&lA[m + 16][0], half);
        v16bf bfr;
#pragma unroll
        for (int j = 0; j < 8; ++j) {
            const int kB = (j < 4 ? 2*j : 2*j + 8) + 8*half;          // B 32x16 layout
            bfr[2*j]   = lB[kB][wave*16 + m];
            bfr[2*j+1] = lB[kB+1][wave*16 + m];
        }
        acc0 = __builtin_amdgcn_wmma_f32_16x16x32_bf16(false, afr0, false, bfr,
                                                       (short)0, acc0, false, false);
        acc1 = __builtin_amdgcn_wmma_f32_16x16x32_bf16(false, afr1, false, bfr,
                                                       (short)0, acc1, false, false);
        __syncthreads();
    }

    // --- epilogue: +b1, exact GELU, store bf16 hh[token][f] ---
    const int col  = ft * TILE_N + wave * 16 + m;
    const float bias = b1[e * kF + col];
#pragma unroll
    for (int r = 0; r < 8; ++r) {
        const int lr0 = r + 8*half;             // C layout: VGPR r -> row r / r+8
        if (mt * TILE_M + lr0 < cnt)
            hh[(size_t)ltok[lr0] * kF + col] = (__bf16)gelu_erf(acc0[r] + bias);
        const int lr1 = 16 + r + 8*half;
        if (mt * TILE_M + lr1 < cnt)
            hh[(size_t)ltok[lr1] * kF + col] = (__bf16)gelu_erf(acc1[r] + bias);
    }
}

__global__ __launch_bounds__(256)
void moe_gemm2(const __bf16* __restrict__ hh, const float* __restrict__ w2,
               const float* __restrict__ b2, const int* __restrict__ counts,
               const int* __restrict__ perm, const float* __restrict__ wgt,
               float* __restrict__ out)
{
    const int e  = blockIdx.x;
    const int dt = blockIdx.y;          // D tile (D/128)
    const int mt = blockIdx.z;
    const int cnt = counts[e];
    if (mt * TILE_M >= cnt) return;

    __shared__ __attribute__((aligned(16))) __bf16 lA[TILE_M][LA_S];
    __shared__ __attribute__((aligned(16))) __bf16 lB[TILE_K][LB_S];
    __shared__ int ltok[TILE_M];

    const int tid  = threadIdx.x;
    const int wave = tid >> 5;
    const int lane = tid & 31;
    const int half = lane >> 4;
    const int m    = lane & 15;

    if (tid < TILE_M) {
        int idx = mt * TILE_M + tid;
        if (idx >= cnt) idx = cnt - 1;
        ltok[tid] = perm[e * kN + idx];
    }
    __syncthreads();

    const int arow = tid >> 3, acol = (tid & 7) * 4;       // A: 32x32 bf16, 8 B/thread
    const int brow = tid >> 5, bcol = (tid & 31) * 4;
    const __bf16* asrc = hh + (size_t)ltok[arow] * kF + acol;
    const float*  bsrc = w2 + (size_t)e * kF * kD + (size_t)brow * kD + dt * TILE_N + bcol;
    // LDS byte address for this thread's async destination (constant over K)
    const unsigned lds_a = (unsigned)(size_t)&lA[arow][acol];

    v8f acc0 = {0.f,0.f,0.f,0.f,0.f,0.f,0.f,0.f};
    v8f acc1 = acc0;

    for (int k0 = 0; k0 < kF; k0 += TILE_K) {
        // --- stage A: async DMA global(bf16) -> LDS, no VGPR round-trip ---
        {
            const unsigned long long ga = (unsigned long long)(asrc + k0);
            asm volatile("global_load_async_to_lds_b64 %0, %1, off"
                         :: "v"(lds_a), "v"(ga)
                         : "memory");
        }
        // --- stage B: 32x128 fp32 -> bf16 from w2[e] ---
#pragma unroll
        for (int rr = 0; rr < TILE_K; rr += 8) {
            const float4 v = *(const float4*)(bsrc + (size_t)(k0 + rr) * kD);
            lB[brow + rr][bcol+0] = (__bf16)v.x;  lB[brow + rr][bcol+1] = (__bf16)v.y;
            lB[brow + rr][bcol+2] = (__bf16)v.z;  lB[brow + rr][bcol+3] = (__bf16)v.w;
        }
        if (k0 + TILE_K < kF) {
            __builtin_prefetch(asrc + k0 + TILE_K, 0, 3);
            __builtin_prefetch(bsrc + (size_t)(k0 + TILE_K) * kD, 0, 3);
        }
        asm volatile("s_wait_asynccnt 0x0" ::: "memory");
        __syncthreads();

        const v16bf afr0 = frag_a(&lA[m][0],      half);
        const v16bf afr1 = frag_a(&lA[m + 16][0], half);
        v16bf bfr;
#pragma unroll
        for (int j = 0; j < 8; ++j) {
            const int kB = (j < 4 ? 2*j : 2*j + 8) + 8*half;
            bfr[2*j]   = lB[kB][wave*16 + m];
            bfr[2*j+1] = lB[kB+1][wave*16 + m];
        }
        acc0 = __builtin_amdgcn_wmma_f32_16x16x32_bf16(false, afr0, false, bfr,
                                                       (short)0, acc0, false, false);
        acc1 = __builtin_amdgcn_wmma_f32_16x16x32_bf16(false, afr1, false, bfr,
                                                       (short)0, acc1, false, false);
        __syncthreads();
    }

    // --- epilogue: +b2, * sigmoid gate, fp32 scatter to out[token][d] ---
    const int col  = dt * TILE_N + wave * 16 + m;
    const float bias = b2[e * kD + col];
#pragma unroll
    for (int r = 0; r < 8; ++r) {
        const int lr0 = r + 8*half;
        if (mt * TILE_M + lr0 < cnt) {
            const int tok = ltok[lr0];
            out[(size_t)tok * kD + col] = (acc0[r] + bias) * wgt[tok];
        }
        const int lr1 = 16 + r + 8*half;
        if (mt * TILE_M + lr1 < cnt) {
            const int tok = ltok[lr1];
            out[(size_t)tok * kD + col] = (acc1[r] + bias) * wgt[tok];
        }
    }
}

// ---------------------------------------------------------------------------
// Host launcher (graph-capture safe: only async stream ops, scratch in d_ws)
// ---------------------------------------------------------------------------
extern "C" void kernel_launch(void* const* d_in, const int* in_sizes, int n_in,
                              void* d_out, int out_size, void* d_ws, size_t ws_size,
                              hipStream_t stream)
{
    (void)in_sizes; (void)n_in; (void)out_size; (void)ws_size;
    const float* x  = (const float*)d_in[0];   // h       (B,S,D)
    const float* rw = (const float*)d_in[1];   // router_w (D,E)
    const float* w1 = (const float*)d_in[2];   // (E,D,F)
    const float* b1 = (const float*)d_in[3];   // (E,F)
    const float* w2 = (const float*)d_in[4];   // (E,F,D)
    const float* b2 = (const float*)d_in[5];   // (E,D)
    float* out = (float*)d_out;

    char* ws = (char*)d_ws;
    int*    counts = (int*)ws;                                   // 64 B (pad to 256)
    int*    perm   = (int*)(ws + 256);                           // E*N*4   = 256 KB
    float*  wgt    = (float*)(ws + 256 + kE * kN * 4);           // N*4     = 16 KB
    __bf16* hh     = (__bf16*)(ws + 256 + kE * kN * 4 + kN * 4); // N*F*2   = 16 MB

    hipMemsetAsync(counts, 0, 64, stream);
    moe_router<<<kN / 8, 256, 0, stream>>>(x, rw, wgt, counts, perm);

    dim3 g1(kE, kF / TILE_N, kN / TILE_M);
    moe_gemm1<<<g1, 256, 0, stream>>>(x, w1, b1, counts, perm, hh);

    dim3 g2(kE, kD / TILE_N, kN / TILE_M);
    moe_gemm2<<<g2, 256, 0, stream>>>(hh, w2, b2, counts, perm, wgt, out);
}
